// ContrastiveLoss_26448408609141
// MI455X (gfx1250) — matrix operand order repaired and checked by reading, not compiled, not run on hardware
//
#include <hip/hip_runtime.h>

// ---- problem constants (match reference) ----
#define TEMP_INV 10.0f      // 1/0.1
#define EPSV     1e-8f
#define NB   8              // B
#define NC   256            // C
#define NT   1024           // T
#define NM   512            // M (masked per batch)
#define NK   100            // K negatives

typedef __attribute__((ext_vector_type(16))) __bf16        v16bf;
typedef __attribute__((ext_vector_type(8)))  float         v8f;
typedef __attribute__((ext_vector_type(4)))  float         f4;
typedef __attribute__((ext_vector_type(4)))  unsigned int  u4;

__device__ __forceinline__ unsigned short f2bf_rne(float f) {
    union { float f; unsigned u; } x; x.f = f;
    unsigned u = x.u;
    u += 0x7FFFu + ((u >> 16) & 1u);          // round-to-nearest-even
    return (unsigned short)(u >> 16);
}

// pack two fp32 bit patterns into one dword of truncated bf16 (lo, hi)
__device__ __forceinline__ unsigned packbf(unsigned lo, unsigned hi) {
    return (lo >> 16) | (hi & 0xFFFF0000u);
}

// ---------------------------------------------------------------------------
// Kernel 1: build per-batch list of masked t indices (== stable argsort of
// where(mask,0,1) taken first M), and zero the output accumulator.
// ---------------------------------------------------------------------------
__global__ void cl_prep_kernel(const unsigned char* __restrict__ mask,
                               int* __restrict__ idx_out,
                               float* __restrict__ out) {
    const int b = blockIdx.x;
    if (b == 0 && threadIdx.x == 0) out[0] = 0.0f;
    if (threadIdx.x == 0) {
        const unsigned char* mrow = mask + (size_t)b * NT;
        int n = 0;
        for (int t = 0; t < NT && n < NM; ++t)
            if (mrow[t]) idx_out[b * NM + (n++)] = t;
    }
}

// ---------------------------------------------------------------------------
// Kernel 2: one workgroup per (b, m). 8 waves:
//   all waves : gather ctx/pos column t (fp32, L2-resident), reduce pos
//               cosine, and stash ctx as bf16 in LDS (one-time conversion)
//   waves 0..6: 16-wide k-tile; 8x v_wmma_f32_16x16x32_bf16 over C=256 with
//               branch-free NT b128 streaming of negatives; fp32 sum-of-
//               squares for the negative norms accumulated alongside
//   thread 0  : 101-way logsumexp, atomicAdd scaled partial into out
// ---------------------------------------------------------------------------
__global__ void __launch_bounds__(256)
cl_loss_kernel(const float* __restrict__ context,
               const float* __restrict__ positive,
               const float* __restrict__ negatives,
               const int*   __restrict__ idx,
               float* __restrict__ out) {
    __shared__ unsigned short s_ctxbf[NC];   // ctx column as bf16 (512 B)
    __shared__ float s_neg[112];             // neg logits (padded to 7*16)
    __shared__ float s_red[3];               // dot, |ctx|^2, |pos|^2
    __shared__ float s_cn, s_poslogit;

    const int m    = blockIdx.x;
    const int b    = blockIdx.y;
    const int tid  = threadIdx.x;
    const int lane = tid & 31;
    const int wid  = tid >> 5;
    const int t    = idx[b * NM + m];

    if (tid < 3) s_red[tid] = 0.0f;
    __syncthreads();

    // ---- gather one column of context/positive: stride-T loads, L2 hits ----
    const size_t colbase = ((size_t)b * NC + tid) * NT + t;
    const float cv = context[colbase];
    const float pv = positive[colbase];
    s_ctxbf[tid] = f2bf_rne(cv);             // one-time bf16 conversion

    // wave-level tree reduction, one LDS atomic per wave
    float rdot = cv * pv, rc2 = cv * cv, rp2 = pv * pv;
    #pragma unroll
    for (int s = 16; s > 0; s >>= 1) {
        rdot += __shfl_xor(rdot, s);
        rc2  += __shfl_xor(rc2,  s);
        rp2  += __shfl_xor(rp2,  s);
    }
    if (lane == 0) {
        atomicAdd(&s_red[0], rdot);
        atomicAdd(&s_red[1], rc2);
        atomicAdd(&s_red[2], rp2);
    }
    __syncthreads();

    if (tid == 0) {
        const float cn = fmaxf(sqrtf(s_red[1]), EPSV);
        const float pn = fmaxf(sqrtf(s_red[2]), EPSV);
        s_cn = cn;
        s_poslogit = (s_red[0] / (cn * pn)) * TEMP_INV;
    }
    __syncthreads();

    // ---- negative similarities via WMMA (waves 0..6, wave-uniform branch) ----
    if (wid < 7) {
        const int  kbase  = wid * 16;
        const int  krow   = kbase + (lane & 15);
        const bool kvalid = (krow < NK);
        const int  kclmp  = kvalid ? krow : (NK - 1);  // branch-free in-bounds
        const int  hi     = lane >> 4;                 // 0: K 0..15, 1: K 16..31
        const int  kofs   = hi ? 16 : 0;
        const float* nrow =
            negatives + ((((size_t)b * NM + m) * NK + (size_t)kclmp) << 8);

        v8f acc = {0.f, 0.f, 0.f, 0.f, 0.f, 0.f, 0.f, 0.f};
        float ss = 0.0f;                               // fp32 sum of squares

        #pragma unroll
        for (int cc = 0; cc < 8; ++cc) {
            const int c0 = cc * 32;

            // B fragment: column N = lane&15 (= row k of negatives), this lane
            // holds 16 consecutive C-values -> 4 coalesced NT b128 loads
            const f4* p = (const f4*)(nrow + c0 + hi * 16);
            const f4 q0 = __builtin_nontemporal_load(p + 0);
            const f4 q1 = __builtin_nontemporal_load(p + 1);
            const f4 q2 = __builtin_nontemporal_load(p + 2);
            const f4 q3 = __builtin_nontemporal_load(p + 3);

            // fp32 norm accumulation (plain VALU, co-executes with WMMA)
            ss += q0.x*q0.x + q0.y*q0.y + q0.z*q0.z + q0.w*q0.w;
            ss += q1.x*q1.x + q1.y*q1.y + q1.z*q1.z + q1.w*q1.w;
            ss += q2.x*q2.x + q2.y*q2.y + q2.z*q2.z + q2.w*q2.w;
            ss += q3.x*q3.x + q3.y*q3.y + q3.z*q3.z + q3.w*q3.w;

            // pack to bf16 by truncation: 2 values per dword
            union { f4 f; u4 u; } b0, b1, b2, b3;
            b0.f = q0; b1.f = q1; b2.f = q2; b3.f = q3;
            union { unsigned u[8]; v16bf v; } bB;
            bB.u[0] = packbf(b0.u.x, b0.u.y);
            bB.u[1] = packbf(b0.u.z, b0.u.w);
            bB.u[2] = packbf(b1.u.x, b1.u.y);
            bB.u[3] = packbf(b1.u.z, b1.u.w);
            bB.u[4] = packbf(b2.u.x, b2.u.y);
            bB.u[5] = packbf(b2.u.z, b2.u.w);
            bB.u[6] = packbf(b3.u.x, b3.u.y);
            bB.u[7] = packbf(b3.u.z, b3.u.w);

            // A fragment: ctx broadcast into all 16 rows, fetched pre-converted
            // from LDS. Layout (matches B): lanes 0-15 hold K{0..15},
            // lanes 16-31 hold K{16..31} -> two 16-byte LDS reads.
            union { u4 q[2]; v16bf v; } aA;
            aA.q[0] = *(const u4*)&s_ctxbf[c0 + kofs];        // K 0..7  / 16..23
            aA.q[1] = *(const u4*)&s_ctxbf[c0 + kofs + 8];    // K 8..15 / 24..31

            acc = __builtin_amdgcn_wmma_f32_16x16x32_bf16(
                    false, aA.v, false, bB.v, (short)0, acc, false, false);
        }

        // fold the two K-halves of the norm (lanes L and L+16 share row k)
        ss += __shfl_xor(ss, 16);

        if ((lane < 16) && kvalid) {
            const float y  = acc[0];                   // D[0, lane] = dot(ctx, neg[k])
            const float nk = fmaxf(sqrtf(ss), EPSV);
            s_neg[krow] = (y / (s_cn * nk)) * TEMP_INV;
        }
    }
    __syncthreads();

    // ---- logsumexp over 101 logits, accumulate scaled partial ----
    if (tid == 0) {
        const float l0 = s_poslogit;
        float mx = l0;
        for (int k = 0; k < NK; ++k) mx = fmaxf(mx, s_neg[k]);
        float se = __expf(l0 - mx);
        for (int k = 0; k < NK; ++k) se += __expf(s_neg[k] - mx);
        const float lse = mx + __logf(se);
        atomicAdd(out, (lse - l0) * (1.0f / (float)(NB * NM)));
    }
}

// ---------------------------------------------------------------------------
extern "C" void kernel_launch(void* const* d_in, const int* in_sizes, int n_in,
                              void* d_out, int out_size, void* d_ws, size_t ws_size,
                              hipStream_t stream) {
    const float*         context   = (const float*)d_in[0];
    const float*         positive  = (const float*)d_in[1];
    const float*         negatives = (const float*)d_in[2];
    const unsigned char* mask      = (const unsigned char*)d_in[3];
    // d_in[4] = num_masked (known 512)
    float* out    = (float*)d_out;
    int*   idxbuf = (int*)d_ws;    // B*M ints = 16 KB of scratch

    cl_prep_kernel<<<dim3(NB), dim3(32), 0, stream>>>(mask, idxbuf, out);
    cl_loss_kernel<<<dim3(NM, NB), dim3(256), 0, stream>>>(
        context, positive, negatives, idxbuf, out);
}
